// VariableDeformableAttentionBlock_76347338654193
// MI455X (gfx1250) — matrix operand order repaired and checked by reading, not compiled, not run on hardware
//
#include <hip/hip_runtime.h>
#include <hip/hip_bf16.h>
#include <math.h>

typedef __attribute__((ext_vector_type(16))) _Float16 v16h;
typedef __attribute__((ext_vector_type(8)))  float    v8f;
typedef __attribute__((ext_vector_type(4)))  unsigned int u32x4;
typedef __attribute__((ext_vector_type(8)))  int      i32x8;
typedef __attribute__((ext_vector_type(4)))  int      i32x4;

// ---------------- TDM (Tensor Data Mover) support, toolchain-gated ----------------
#if defined(__has_builtin)
#  if __has_builtin(__builtin_amdgcn_tensor_load_to_lds)
#    define HAVE_TDM 1
#  endif
#endif
#ifndef HAVE_TDM
#  define HAVE_TDM 0
#endif
#if __has_include(<hip/amd_detail/amd_gfx1250_TDM.h>)
#  define TDM_6ARG 1
#else
#  define TDM_6ARG 0
#endif

__device__ __forceinline__ unsigned int lds_off_of(const void* p) {
  // Generic (flat) shared pointer: low 32 bits are the LDS byte offset (ISA aperture rule).
  return (unsigned int)(unsigned long long)(uintptr_t)p;
}

#if HAVE_TDM
// 1-D contiguous tile load: nelem elements of (1<<dsz_log) bytes, nelem < 65536.
// D# group0: [1:0]=count=1 | lds_addr[63:32] | global_addr[120:64] | type=2 @ [127:126]
// D# group1: data_size[17:16]; tensor_dim0[79:48]; tile_dim0[127:112]; dim0_stride[207:160]
__device__ __forceinline__ void tdm_load_1d(unsigned int lds_byte, const void* gptr,
                                            unsigned int nelem, unsigned int dsz_log) {
  unsigned long long ga = (unsigned long long)(uintptr_t)gptr;
  u32x4 g0;
  g0[0] = 1u;                                   // count=1, user mode, no gather
  g0[1] = lds_byte;                             // lds_addr
  g0[2] = (unsigned int)ga;                     // global_addr[31:0]
  g0[3] = (unsigned int)((ga >> 32) & 0x1FFFFFFu) | (2u << 30);  // addr[56:32] | type=2
  i32x8 g1;
  g1[0] = (int)(dsz_log << 16);                 // workgroup_mask=0, data_size
  g1[1] = (int)((nelem & 0xFFFFu) << 16);       // tensor_dim0[15:0] -> bits 63:48
  g1[2] = (int)((nelem >> 16) & 0xFFFFu);       // tensor_dim0[31:16]; tensor_dim1=0
  g1[3] = (int)((nelem & 0xFFFFu) << 16);       // tile_dim0 -> bits 127:112
  g1[4] = 0;                                    // tile_dim1=0 (1-D), tile_dim2=0
  g1[5] = (int)nelem;                           // tensor_dim0_stride (unused for 1-D)
  g1[6] = 0;
  g1[7] = 0;
  i32x4 z4 = {0, 0, 0, 0};
#if TDM_6ARG
  i32x8 z8 = {0, 0, 0, 0, 0, 0, 0, 0};
  __builtin_amdgcn_tensor_load_to_lds(g0, g1, z4, z4, z8, 0);
#else
  __builtin_amdgcn_tensor_load_to_lds(g0, g1, z4, z4, 0);
#endif
}
#endif  // HAVE_TDM

// ---------- WMMA fragment helpers (CDNA5 16x16x32 f16, wave32) ----------
__device__ __forceinline__ v16h lds_frag_a(const _Float16* base, int ld, int row0, int k0, int lane) {
  v16h a;
  int m  = row0 + (lane & 15);
  int ks = k0 + ((lane >> 4) << 3);
  const _Float16* p0 = base + m * ld + ks;
#pragma unroll
  for (int t = 0; t < 8; ++t) { a[t] = p0[t]; a[8 + t] = p0[16 + t]; }
  return a;
}

__device__ __forceinline__ v16h lds_frag_a_guard(const _Float16* base, int ld, int row0, int k0, int lane, int nrows) {
  v16h a;
  int m  = row0 + (lane & 15);
  if (m < 0 || m >= nrows) {
#pragma unroll
    for (int t = 0; t < 16; ++t) a[t] = (_Float16)0.f;
    return a;
  }
  int ks = k0 + ((lane >> 4) << 3);
  const _Float16* p0 = base + m * ld + ks;
#pragma unroll
  for (int t = 0; t < 8; ++t) { a[t] = p0[t]; a[8 + t] = p0[16 + t]; }
  return a;
}

__device__ __forceinline__ v16h ld_frag_b(const _Float16* w, int ldw, int n0, int k0, int lane) {
  int nn = n0 + (lane & 15);
  int ks = k0 + ((lane >> 4) << 3);
  const _Float16* p = w + nn * ldw + ks;
  v16h b;
#pragma unroll
  for (int t = 0; t < 8; ++t) { b[t] = p[t]; b[8 + t] = p[16 + t]; }
  return b;
}

__device__ __forceinline__ v8f wmma_f16(v16h a, v16h b, v8f c) {
  return __builtin_amdgcn_wmma_f32_16x16x32_f16(false, a, false, b, (short)0, c, false, false);
}

// ---------- Kernel 0: weight folding & f16 repack ----------
// consts (floats): [0:128) wkr=rowsum(WK)/sqrt(D), [128:256) wvr, [256:384) u, [384:512) c,
//                  [512:640) g, [640:768) hvec, [768:770) cb
__global__ __launch_bounds__(256) void vdab_precompute(
    const float* WQ_w, const float* WK_w, const float* WV_w, const float* WV_b,
    const float* Wi_w, const float* Wi_b, const float* Wv_w, const float* Wv_b,
    const float* conv1_w, const float* conv1_b, const float* conv2_w, const float* conv2_b,
    _Float16* wqh, _Float16* wc, float* consts) {
  int tid = threadIdx.x;
  float* wkr  = consts;       float* wvr  = consts + 128;
  float* u    = consts + 256; float* cvec = consts + 384;
  float* g    = consts + 512; float* hv   = consts + 640;
  float* cb   = consts + 768;

  for (int i = tid; i < 128 * 128; i += 256) wqh[i] = (_Float16)WQ_w[i];
  for (int i = tid; i < 9 * 128 * 128; i += 256) {
    int tap = i >> 14; int rem = i & 16383; int oc = rem >> 7; int ic = rem & 127;
    wc[i] = (_Float16)conv1_w[(oc * 128 + ic) * 9 + tap];
  }
  if (tid < 128) {
    float sk = 0.f, sv = 0.f;
    for (int d = 0; d < 128; ++d) { sk += WK_w[tid * 128 + d]; sv += WV_w[tid * 128 + d]; }
    wkr[tid] = sk * 0.08838834764831845f; // 1/sqrt(128)
    wvr[tid] = sv;
  }
  if (tid < 2) {
    float s = conv2_b[tid];
    for (int oc = 0; oc < 128; ++oc) s += conv1_b[oc] * conv2_w[tid * 128 + oc];
    cb[tid] = s;
  }
  __syncthreads();
  if (tid < 128) {
    float su = 0.f, sc = 0.f;
    for (int d = 0; d < 128; ++d) { float w = Wi_w[tid * 128 + d]; su += w * wvr[d]; sc += w * WV_b[d]; }
    u[tid] = su; cvec[tid] = sc + Wi_b[tid];
  }
  __syncthreads();
  if (tid < 128) {
    float sg = 0.f, sh = 0.f;
    for (int d = 0; d < 128; ++d) { float w = Wv_w[tid * 128 + d]; sg += w * u[d]; sh += w * cvec[d]; }
    g[tid] = sg; hv[tid] = sh + Wv_b[tid];
  }
}

// ---------- Kernel 1: Q projection (WMMA) + qk1 logit scalars ----------
__global__ __launch_bounds__(256) void vdab_qproj(
    const float* x, const _Float16* wqh, const float* WQ_b,
    const float* consts, _Float16* qp16, float* qk1) {
  __shared__ _Float16 Z[64 * 128];      // 16 KB (f16 A-operand)
#if HAVE_TDM
  __shared__ float Xs[64 * 128];        // 32 KB TDM landing buffer
#endif
  __shared__ float qkacc[64];
  int tid = threadIdx.x;
  int b = blockIdx.x / 65, n = blockIdx.x % 65;
  const float* wkr = consts;
  const float* src = x + ((size_t)b * 4096 + (size_t)n * 32) * 128;

  if (tid < 64) qkacc[tid] = 0.f;
#if HAVE_TDM
  // TDM: DMA the 32 KB f32 window into LDS, then convert to f16 in LDS.
  if (tid == 0) {
    tdm_load_1d(lds_off_of(Xs), src, 64 * 128, 2 /*4-byte*/);
    __builtin_amdgcn_s_wait_tensorcnt(0);
  }
  __syncthreads();
  for (int i = tid; i < 2048; i += 256) {
    float4 v = ((const float4*)Xs)[i];
    int o = i * 4;
    Z[o] = (_Float16)v.x; Z[o + 1] = (_Float16)v.y; Z[o + 2] = (_Float16)v.z; Z[o + 3] = (_Float16)v.w;
  }
#else
  for (int i = tid; i < 2048; i += 256) {
    float4 v = ((const float4*)src)[i];
    int o = i * 4;
    Z[o] = (_Float16)v.x; Z[o + 1] = (_Float16)v.y; Z[o + 2] = (_Float16)v.z; Z[o + 3] = (_Float16)v.w;
  }
#endif
  __syncthreads();

  int wave = tid >> 5, lane = tid & 31;
  int rb = wave >> 1, ch = wave & 1;
  v8f acc[4];
#pragma unroll
  for (int t = 0; t < 4; ++t)
#pragma unroll
    for (int r = 0; r < 8; ++r) acc[t][r] = 0.f;

#pragma unroll
  for (int kc = 0; kc < 4; ++kc) {
    v16h a = lds_frag_a(Z, 128, rb * 16, kc * 32, lane);
#pragma unroll
    for (int t = 0; t < 4; ++t) {
      v16h bf = ld_frag_b(wqh, 128, ch * 64 + t * 16, kc * 32, lane);
      acc[t] = wmma_f16(a, bf, acc[t]);
    }
  }

  size_t base = (size_t)blockIdx.x * 8192;
#pragma unroll
  for (int t = 0; t < 4; ++t) {
    int oc = ch * 64 + t * 16 + (lane & 15);
    float wq = wkr[oc], bq = WQ_b[oc];
#pragma unroll
    for (int r = 0; r < 8; ++r) {
      int m = rb * 16 + r + ((lane >> 4) << 3);
      float val = acc[t][r] + bq;
      qp16[base + (size_t)m * 128 + oc] = (_Float16)val;
      atomicAdd(&qkacc[m], val * wq);
    }
  }
  __syncthreads();
  if (tid < 64) qk1[(size_t)blockIdx.x * 64 + tid] = qkacc[tid];
}

// ---------- Kernel 2: conv1 (implicit GEMM, WMMA) fused with conv2 + tanh -> delta ----------
__global__ __launch_bounds__(256) void vdab_conv_offset(
    const _Float16* qp16, const _Float16* wc, const float* conv2_w,
    const float* consts, const float* alpha_p, float* delta) {
  __shared__ _Float16 Qw[3 * 64 * 128];   // 48 KB: windows nw-1, nw, nw+1
  __shared__ float off2[64][2];
  int tid = threadIdx.x;
  int b = blockIdx.x >> 6, nw = blockIdx.x & 63;

  if (tid < 64) { off2[tid][0] = 0.f; off2[tid][1] = 0.f; }
#if HAVE_TDM
  // TDM: DMA up to three 16 KB f16 windows straight into LDS.
  if (tid == 0) {
#pragma unroll
    for (int w3 = 0; w3 < 3; ++w3) {
      int srcwin = nw - 1 + w3;
      if (srcwin >= 0)
        tdm_load_1d(lds_off_of(Qw) + (unsigned int)w3 * 16384u,
                    qp16 + (size_t)(b * 65 + srcwin) * 8192, 64 * 128, 1 /*2-byte*/);
    }
    __builtin_amdgcn_s_wait_tensorcnt(0);
  }
  if (nw == 0) {  // window 0 is the n=-1 zero pad; disjoint from TDM targets
    uint4 zz; zz.x = zz.y = zz.z = zz.w = 0u;
    for (int c = tid; c < 1024; c += 256) ((uint4*)Qw)[c] = zz;
  }
#else
  uint4* dst = (uint4*)Qw;
  for (int c = tid; c < 3072; c += 256) {
    int w3 = c >> 10;
    int srcwin = nw - 1 + w3;
    uint4 v;
    if (srcwin < 0) { v.x = v.y = v.z = v.w = 0u; }
    else v = ((const uint4*)(qp16 + (size_t)(b * 65 + srcwin) * 8192))[c & 1023];
    dst[c] = v;
  }
#endif
  __syncthreads();

  int wave = tid >> 5, lane = tid & 31;
  int rb = wave >> 1, ch = wave & 1;
  v8f acc[4];
#pragma unroll
  for (int t = 0; t < 4; ++t)
#pragma unroll
    for (int r = 0; r < 8; ++r) acc[t][r] = 0.f;

  for (int tap = 0; tap < 9; ++tap) {
    int w3 = tap / 3;
    int dl = tap % 3 - 1;
    const _Float16* abase = Qw + w3 * 8192;
    const _Float16* wt = wc + tap * 16384;
    __builtin_prefetch(wt + (size_t)lane * 64, 0, 0);
#pragma unroll
    for (int kc = 0; kc < 4; ++kc) {
      v16h a = lds_frag_a_guard(abase, 128, rb * 16 + dl, kc * 32, lane, 64);
#pragma unroll
      for (int t = 0; t < 4; ++t) {
        v16h bf = ld_frag_b(wt, 128, ch * 64 + t * 16, kc * 32, lane);
        acc[t] = wmma_f16(a, bf, acc[t]);
      }
    }
  }

  // conv2 (1x1, 128 -> 2) straight from accumulators via LDS float atomics
  const float* w20 = conv2_w;
  const float* w21 = conv2_w + 128;
#pragma unroll
  for (int t = 0; t < 4; ++t) {
    int oc = ch * 64 + t * 16 + (lane & 15);
    float c0 = w20[oc], c1 = w21[oc];
#pragma unroll
    for (int r = 0; r < 8; ++r) {
      int m = rb * 16 + r + ((lane >> 4) << 3);
      float v = acc[t][r];
      atomicAdd(&off2[m][0], v * c0);
      atomicAdd(&off2[m][1], v * c1);
    }
  }
  __syncthreads();
  if (tid < 64) {
    float al = alpha_p[0];
    const float* cb = consts + 768;
    float d0 = al * tanhf(off2[tid][0] + cb[0]);
    float d1 = al * tanhf(off2[tid][1] + cb[1]);
    size_t o = ((size_t)blockIdx.x * 64 + tid) * 2;
    delta[o] = d0; delta[o + 1] = d1;
  }
}

// ---------- Kernel 3: bilinear sampling + 64-wide softmax + rank-1 output ----------
__global__ __launch_bounds__(256) void vdab_attn_out(
    const float* x, const float* qk1, const float* delta,
    const float* consts, float* out) {
  __shared__ float smp[64], tvals[64], qkl[64];
  __shared__ float gsh[128], hsh[128];
  int tid = threadIdx.x;
  int b = blockIdx.x >> 6, nw = blockIdx.x & 63;

  if (tid < 128) { gsh[tid] = consts[512 + tid]; hsh[tid] = consts[640 + tid]; }
  if (tid < 64) {
    size_t dof = ((size_t)blockIdx.x * 64 + tid) * 2;
    float d0 = delta[dof], d1 = delta[dof + 1];
    float xc = (float)tid + d0, yc = d1;
    int x0 = (int)truncf(xc); int x1 = min(x0 + 1, 63);
    int y0 = (int)truncf(yc); int y1 = min(y0 + 1, 127);
    const float* zb = x + ((size_t)b * 4096 + (size_t)nw * 32) * 128;
    auto samp = [&](int xi, int yi) -> float {
      bool valid = (xi >= 0) && (xi < 64) && (yi >= 0) && (yi < 128);
      int xic = min(max(xi, 0), 63), yic = min(max(yi, 0), 127);
      float v = zb[xic * 128 + yic];
      return valid ? v : 0.f;
    };
    float Ia = samp(x0, y0), Ib = samp(x1, y0), Ic = samp(x0, y1), Id = samp(x1, y1);
    float x0f = (float)x0, x1f = (float)x1, y0f = (float)y0, y1f = (float)y1;
    smp[tid] = (x1f - xc) * (y1f - yc) * Ia + (xc - x0f) * (y1f - yc) * Ib +
               (x1f - xc) * (yc - y0f) * Ic + (xc - x0f) * (yc - y0f) * Id;
    qkl[tid] = qk1[(size_t)(b * 65 + nw) * 64 + tid];
  }
  __syncthreads();
  if (tid < 64) {
    float q = qkl[tid];
    float mx = -INFINITY;
    for (int j = 0; j < 64; ++j) mx = fmaxf(mx, q * smp[j]);
    float se = 0.f, sw = 0.f;
    for (int j = 0; j < 64; ++j) {
      float e = expf(q * smp[j] - mx);
      se += e; sw += e * smp[j];
    }
    tvals[tid] = sw / se;
  }
  __syncthreads();
  float4* outb = (float4*)(out + ((size_t)b * 4096 + (size_t)nw * 64) * 128);
  const float4* g4 = (const float4*)gsh;
  const float4* h4 = (const float4*)hsh;
  for (int idx = tid; idx < 2048; idx += 256) {
    int i = idx >> 5, o4 = idx & 31;
    float tt = tvals[i];
    float4 gv = g4[o4], hv = h4[o4];
    float4 r;
    r.x = tt * gv.x + hv.x; r.y = tt * gv.y + hv.y;
    r.z = tt * gv.z + hv.z; r.w = tt * gv.w + hv.w;
    outb[idx] = r;
  }
}

extern "C" void kernel_launch(void* const* d_in, const int* in_sizes, int n_in,
                              void* d_out, int out_size, void* d_ws, size_t ws_size,
                              hipStream_t stream) {
  const float* x       = (const float*)d_in[0];
  const float* WQ_w    = (const float*)d_in[1];
  const float* WQ_b    = (const float*)d_in[2];
  const float* WK_w    = (const float*)d_in[3];
  const float* WK_b    = (const float*)d_in[4];
  const float* WV_w    = (const float*)d_in[5];
  const float* WV_b    = (const float*)d_in[6];
  const float* Wi_w    = (const float*)d_in[7];
  const float* Wi_b    = (const float*)d_in[8];
  const float* Wv_w    = (const float*)d_in[9];
  const float* Wv_b    = (const float*)d_in[10];
  const float* conv1_w = (const float*)d_in[11];
  const float* conv1_b = (const float*)d_in[12];
  const float* conv2_w = (const float*)d_in[13];
  const float* conv2_b = (const float*)d_in[14];
  const float* alpha   = (const float*)d_in[15];

  char* ws = (char*)d_ws;
  size_t off = 0;
  auto take = [&](size_t bytes) -> void* {
    off = (off + 255) & ~(size_t)255;
    void* p = ws + off;
    off += bytes;
    return p;
  };
  _Float16* qp16   = (_Float16*)take((size_t)16 * 65 * 64 * 128 * 2);  // ~17 MB
  float*    qk1f   = (float*)take((size_t)16 * 65 * 64 * 4);
  float*    deltf  = (float*)take((size_t)16 * 64 * 64 * 2 * 4);
  _Float16* wqh    = (_Float16*)take((size_t)128 * 128 * 2);
  _Float16* wc     = (_Float16*)take((size_t)9 * 128 * 128 * 2);
  float*    consts = (float*)take((size_t)(6 * 128 + 2) * 4);

  vdab_precompute<<<1, 256, 0, stream>>>(WQ_w, WK_w, WV_w, WV_b, Wi_w, Wi_b,
                                         Wv_w, Wv_b, conv1_w, conv1_b,
                                         conv2_w, conv2_b, wqh, wc, consts);
  vdab_qproj<<<16 * 65, 256, 0, stream>>>(x, wqh, WQ_b, consts, qp16, qk1f);
  vdab_conv_offset<<<16 * 64, 256, 0, stream>>>(qp16, wc, conv2_w, consts, alpha, deltf);
  vdab_attn_out<<<16 * 64, 256, 0, stream>>>(x, qk1f, deltf, consts, (float*)d_out);
}
